// pwq_layer_9405978378840
// MI455X (gfx1250) — matrix operand order repaired and checked by reading, not compiled, not run on hardware
//
#include <hip/hip_runtime.h>
#include <hip/hip_fp16.h>
#include <math.h>

typedef __attribute__((ext_vector_type(16))) _Float16 v16h;
typedef __attribute__((ext_vector_type(8)))  float    v8f;

#define EPS_F 1.1920929e-7f

// ---- packed f16 weight layout (offsets in halves, each 32B aligned) ----
#define OFF_HW1 0                       // 16x256, K padded to 32 : 16 tiles * 1 step * 512
#define OFF_HW2 (OFF_HW1 + 8192)        // 256x256 : 16 tiles * 8 steps * 512
#define OFF_HW3 (OFF_HW2 + 65536)       // 256x528 : 33 tiles * 8 steps * 512
#define OFF_WW1 (OFF_HW3 + 135168)
#define OFF_WW2 (OFF_WW1 + 8192)
#define OFF_WW3 (OFF_WW2 + 65536)       // 256x512 : 32 tiles * 8 steps * 512
#define PACK_TOTAL (OFF_WW3 + 131072)

// Rows per block: 4 row-groups of 16 share every streamed B fragment,
// quartering L2 weight traffic vs the 16-row baseline.
#define ROWS 64
#define NG   4

// LDS row strides padded by 8 halves (4 dwords) -> 16 rows map to 16
// distinct banks (stride%64 dwords = 4) while keeping 16B alignment so
// A-fragment dword loads still merge into ds_load_b128.
#define XA_LD 40     // 32 + 8
#define H_LD  264    // 256 + 8

// -------------------------------------------------------------------------
// Pack fp32 row-major weights [K x N] into WMMA B-fragment order (f16).
// ISA 7.12.2 (16-bit B, 32x16): lane<16 -> K=32*s+{0..7,16..23} ; lane>=16 -> K=32*s+{8..15,24..31},
// h = 2*vgpr+half = (p & 15). Each lane's 16 halves are contiguous (32B).
// -------------------------------------------------------------------------
__global__ void pack_weights_kernel(const float* __restrict__ src,
                                    _Float16* __restrict__ dst,
                                    int Kvalid, int N, int S, int tiles)
{
    const int total = tiles * S * 512;
    for (int p = blockIdx.x * blockDim.x + threadIdx.x; p < total;
         p += gridDim.x * blockDim.x) {
        const int h    = p & 15;
        const int lane = (p >> 4) & 31;
        const int s    = (p >> 9) % S;
        const int tile = p / (S << 9);
        const int k = (s << 5) + ((lane & 16) ? 8 : 0) + ((h & 8) ? 16 : 0) + (h & 7);
        const int n = (tile << 4) + (lane & 15);
        float v = (k < Kvalid) ? src[k * N + n] : 0.0f;
        dst[p] = (_Float16)v;
    }
}

// -------------------------------------------------------------------------
// A-fragment gather from LDS (row-major f16, row stride ldK halves).
// ISA 7.12.2 16-bit A 16x32: lanes 0-15 -> K {0..7,16..23}; lanes 16-31
// -> K {8..15,24..31}; VGPR v holds pair (even K, K+1).
// -------------------------------------------------------------------------
__device__ __forceinline__ v16h load_a_frag(const _Float16* lds, int ldK,
                                            int s, int lane)
{
    union { v16h v; unsigned int u[8]; } f;
    const int m    = lane & 15;
    const int koff = ((lane >> 4) << 3) + (s << 5);
    const _Float16* row = lds + m * ldK + koff;
#pragma unroll
    for (int r = 0; r < 8; ++r) {
        const int kl = ((r >> 2) << 4) + ((r & 3) << 1);
        f.u[r] = *(const unsigned int*)(row + kl);
    }
    return f.v;
}

__device__ __forceinline__ v16h load_b_frag(const _Float16* pk, int S,
                                            int tile, int s, int lane)
{
    return *(const v16h*)(pk + ((((tile * S + s) << 5) + lane) << 4));
}

// Multi-row-group GEMM tile: one streamed B fragment feeds NG WMMAs.
template <int S>
__device__ __forceinline__ void gemm_tileN(const _Float16* ldsA, int ldK,
                                           const _Float16* pk, int tile,
                                           int lane, v8f acc[NG])
{
#pragma unroll
    for (int s = 0; s < S; ++s) {
        v16h b = load_b_frag(pk, S, tile, s, lane);
#pragma unroll
        for (int g = 0; g < NG; ++g) {
            v16h a = load_a_frag(ldsA + g * 16 * ldK, ldK, s, lane);
            acc[g] = __builtin_amdgcn_wmma_f32_16x16x32_f16(
                false, a, false, b, (short)0, acc[g], false, false);
        }
    }
}

// C/D layout (32-bit 16x16): n = lane&15 (+tile), m = r + 8*(lane>=16).
template <bool RELU>
__device__ __forceinline__ void store_actN(_Float16* out, int ldN, int tile,
                                           int lane, const v8f acc[NG],
                                           const float* __restrict__ bias)
{
    const int n  = (tile << 4) + (lane & 15);
    const int mb = (lane >> 4) << 3;
    const float bv = bias[n];
#pragma unroll
    for (int g = 0; g < NG; ++g) {
#pragma unroll
        for (int r = 0; r < 8; ++r) {
            float v = acc[g][r] + bv;
            if (RELU) v = v > 0.0f ? v : 0.0f;
            out[(g * 16 + mb + r) * ldN + n] = (_Float16)v;
        }
    }
}

// -------------------------------------------------------------------------
// Fused spline-flow kernel: 64 batch rows per block, 8 wave32 per block.
// Static LDS ~209 KB (ISA: up to 320 KB per workgroup) -> 1 block / WGP.
// -------------------------------------------------------------------------
__global__ __launch_bounds__(256)
void spline_flow_kernel(const float* __restrict__ x,
                        const _Float16* __restrict__ wpk,
                        const float* __restrict__ hb1,
                        const float* __restrict__ hb2,
                        const float* __restrict__ hb3,
                        const float* __restrict__ wb1,
                        const float* __restrict__ wb2,
                        const float* __restrict__ wb3,
                        float* __restrict__ out, int nrows)
{
    __shared__ float    xt[ROWS][32];      // raw input tile (fp32)        8 KB
    __shared__ _Float16 xa[ROWS][XA_LD];   // x_A, K padded to 32          5 KB
    __shared__ _Float16 hA[ROWS][H_LD];    // hidden ping                 33 KB
    __shared__ _Float16 hB[ROWS][H_LD];    // hidden pong                 33 KB
    __shared__ _Float16 rawH[ROWS][528];   // h-net layer-3 output        66 KB
    __shared__ _Float16 rawW[ROWS][512];   // w-net layer-3 output        64 KB

    const int tid  = threadIdx.x;
    const int wave = tid >> 5;
    const int lane = tid & 31;
    const int row0 = blockIdx.x * ROWS;

    for (int i = tid; i < ROWS * 32; i += 256) {
        const int r = i >> 5, c = i & 31;
        xt[r][c] = (row0 + r < nrows) ? x[(row0 + r) * 32 + c] : 0.0f;
    }
    __syncthreads();
    for (int i = tid; i < ROWS * XA_LD; i += 256) {
        const int r = i / XA_LD, c = i % XA_LD;
        xa[r][c] = (c < 16) ? (_Float16)xt[r][2 * c] : (_Float16)0.0f;
    }
    __syncthreads();

    // ---------------- h network: 16 -> 256 -> 256 -> 528 ----------------
#pragma unroll
    for (int j = 0; j < 2; ++j) {
        const int tile = wave * 2 + j;
        v8f acc[NG] = {};
        gemm_tileN<1>(&xa[0][0], XA_LD, wpk + OFF_HW1, tile, lane, acc);
        store_actN<true>(&hA[0][0], H_LD, tile, lane, acc, hb1);
    }
    __syncthreads();
#pragma unroll
    for (int j = 0; j < 2; ++j) {
        const int tile = wave * 2 + j;
        v8f acc[NG] = {};
        gemm_tileN<8>(&hA[0][0], H_LD, wpk + OFF_HW2, tile, lane, acc);
        store_actN<true>(&hB[0][0], H_LD, tile, lane, acc, hb2);
    }
    __syncthreads();
    for (int tile = wave; tile < 33; tile += 8) {
        v8f acc[NG] = {};
        gemm_tileN<8>(&hB[0][0], H_LD, wpk + OFF_HW3, tile, lane, acc);
        store_actN<false>(&rawH[0][0], 528, tile, lane, acc, hb3);
    }
    __syncthreads();

    // ---------------- w network: 16 -> 256 -> 256 -> 512 ----------------
#pragma unroll
    for (int j = 0; j < 2; ++j) {
        const int tile = wave * 2 + j;
        v8f acc[NG] = {};
        gemm_tileN<1>(&xa[0][0], XA_LD, wpk + OFF_WW1, tile, lane, acc);
        store_actN<true>(&hA[0][0], H_LD, tile, lane, acc, wb1);
    }
    __syncthreads();
#pragma unroll
    for (int j = 0; j < 2; ++j) {
        const int tile = wave * 2 + j;
        v8f acc[NG] = {};
        gemm_tileN<8>(&hA[0][0], H_LD, wpk + OFF_WW2, tile, lane, acc);
        store_actN<true>(&hB[0][0], H_LD, tile, lane, acc, wb2);
    }
    __syncthreads();
    for (int tile = wave; tile < 32; tile += 8) {
        v8f acc[NG] = {};
        gemm_tileN<8>(&hB[0][0], H_LD, wpk + OFF_WW3, tile, lane, acc);
        store_actN<false>(&rawW[0][0], 512, tile, lane, acc, wb3);
    }
    __syncthreads();

    // ------------- spline evaluation: one thread per (row, dim) ---------
    for (int p = tid; p < ROWS * 16; p += 256) {
        const int r = p >> 4;
        const int d = p & 15;

        float hts[33];
        for (int j = 0; j < 33; ++j) {
            const float rv = (float)rawH[r][d * 33 + j];
            hts[j] = (rv > 20.0f ? rv : log1pf(expf(rv))) + EPS_F;
        }
        float rw[32];
        float mx = -3.0e38f;
        for (int k = 0; k < 32; ++k) {
            rw[k] = (float)rawW[r][d * 32 + k];
            mx = fmaxf(mx, rw[k]);
        }
        float ssum = 0.0f;
        for (int k = 0; k < 32; ++k) { rw[k] = expf(rw[k] - mx); ssum += rw[k]; }
        const float inv = 1.0f / ssum;

        float bins[33];
        bins[0] = 0.0f;
        for (int k = 0; k < 32; ++k) bins[k + 1] = bins[k] + rw[k] * inv;

        float total = 0.0f;
        for (int k = 0; k < 32; ++k)
            total += 0.5f * (hts[k] + hts[k + 1]) * (bins[k + 1] - bins[k]);
        const float invT = 1.0f / total;

        const float xB = xt[r][2 * d + 1];
        int idx = 0;
        for (int j = 1; j < 33; ++j) idx += (xB >= bins[j]) ? 1 : 0;
        idx = idx > 31 ? 31 : idx;

        float cleft = 0.0f;
        for (int k = 0; k < 32; ++k)
            if (k < idx)
                cleft += 0.5f * (hts[k] + hts[k + 1]) * (bins[k + 1] - bins[k]) * invT;

        const float xl = bins[idx], xr = bins[idx + 1];
        const float vl = hts[idx] * invT, vr = hts[idx + 1] * invT;
        const float wb = xr - xl;
        const float alpha = (xB - xl) / (wb + EPS_F);
        const float yB = cleft + 0.5f * alpha * alpha * (vr - vl) * wb
                       + alpha * vl * wb;

        if (row0 + r < nrows) {
            out[(row0 + r) * 32 + 2 * d]     = xt[r][2 * d];
            out[(row0 + r) * 32 + 2 * d + 1] = yB;
        }
    }
}

extern "C" void kernel_launch(void* const* d_in, const int* in_sizes, int n_in,
                              void* d_out, int out_size, void* d_ws, size_t ws_size,
                              hipStream_t stream)
{
    const float* x   = (const float*)d_in[0];
    const float* hW1 = (const float*)d_in[1];
    const float* hb1 = (const float*)d_in[2];
    const float* hW2 = (const float*)d_in[3];
    const float* hb2 = (const float*)d_in[4];
    const float* hW3 = (const float*)d_in[5];
    const float* hb3 = (const float*)d_in[6];
    const float* wW1 = (const float*)d_in[7];
    const float* wb1 = (const float*)d_in[8];
    const float* wW2 = (const float*)d_in[9];
    const float* wb2 = (const float*)d_in[10];
    const float* wW3 = (const float*)d_in[11];
    const float* wb3 = (const float*)d_in[12];
    // d_in[13]/d_in[14] are A_dims/B_dims = even/odd columns (hardcoded).

    _Float16* wpk = (_Float16*)d_ws;

    // Repack fp32 weights -> f16 WMMA B-fragment order in workspace.
    pack_weights_kernel<<<32,  256, 0, stream>>>(hW1, wpk + OFF_HW1, 16,  256, 1, 16);
    pack_weights_kernel<<<256, 256, 0, stream>>>(hW2, wpk + OFF_HW2, 256, 256, 8, 16);
    pack_weights_kernel<<<528, 256, 0, stream>>>(hW3, wpk + OFF_HW3, 256, 528, 8, 33);
    pack_weights_kernel<<<32,  256, 0, stream>>>(wW1, wpk + OFF_WW1, 16,  256, 1, 16);
    pack_weights_kernel<<<256, 256, 0, stream>>>(wW2, wpk + OFF_WW2, 256, 256, 8, 16);
    pack_weights_kernel<<<512, 256, 0, stream>>>(wW3, wpk + OFF_WW3, 256, 512, 8, 32);

    const int nrows = in_sizes[0] / 32;
    const int grid  = (nrows + ROWS - 1) / ROWS;
    spline_flow_kernel<<<grid, 256, 0, stream>>>(x, wpk, hb1, hb2, hb3,
                                                 wb1, wb2, wb3,
                                                 (float*)d_out, nrows);
}
